// FP8LinearDynamic_75368086110391
// MI455X (gfx1250) — 8-bit matrix operand order repaired and checked by reading, compiled, not run on hardware
//
#include <hip/hip_runtime.h>


typedef __attribute__((ext_vector_type(2)))  int   v2i;
typedef __attribute__((ext_vector_type(4)))  int   v4i;
typedef __attribute__((ext_vector_type(8)))  int   v8i;
typedef __attribute__((ext_vector_type(16))) int   v16i;
typedef __attribute__((ext_vector_type(8)))  float v8f;

#define FP8_MAX 448.0f

__device__ __forceinline__ unsigned int cvt1_fp8_sw(float f) {
  float cf = fminf(fmaxf(f, -FP8_MAX), FP8_MAX);
  unsigned int u = __float_as_uint(cf);
  unsigned int s = (u >> 31) << 7;
  int e = (int)((u >> 23) & 0xFF) - 127;
  unsigned int m = (u >> 20) & 0x7u;
  if (e < -6) return s;
  return s | ((unsigned int)(e + 7) << 3) | m;
}

__device__ __forceinline__ unsigned int pack_fp8x4(float a, float b, float c, float d) {
#if __has_builtin(__builtin_amdgcn_cvt_pk_fp8_f32)
  int p = __builtin_amdgcn_cvt_pk_fp8_f32(a, b, 0, false);
  p = __builtin_amdgcn_cvt_pk_fp8_f32(c, d, p, true);
  return (unsigned int)p;
#else
  return cvt1_fp8_sw(a) | (cvt1_fp8_sw(b) << 8) |
         (cvt1_fp8_sw(c) << 16) | (cvt1_fp8_sw(d) << 24);
#endif
}

__global__ __launch_bounds__(256) void fp8_quant_rows(
    const float* __restrict__ x, unsigned char* __restrict__ q,
    float* __restrict__ xscale, int K)
{
  const int lane = threadIdx.x & 31;
  const int wave = threadIdx.x >> 5;
  const long row = (long)blockIdx.x * 8 + wave;

  const float4* xr = (const float4*)(x + row * (long)K);
  float4 v[8];
  float amax = 0.0f;
#pragma unroll
  for (int i = 0; i < 8; ++i) {
    v[i] = xr[i * 32 + lane];
    amax = fmaxf(amax, fmaxf(fmaxf(fabsf(v[i].x), fabsf(v[i].y)),
                             fmaxf(fabsf(v[i].z), fabsf(v[i].w))));
  }
#pragma unroll
  for (int off = 16; off > 0; off >>= 1)
    amax = fmaxf(amax, __shfl_xor(amax, off, 32));
  amax = fmaxf(amax, 1e-12f);

  const float scale = amax * (1.0f / FP8_MAX);
  const float inv   = FP8_MAX / amax;
  if (lane == 0) xscale[row] = scale;

  unsigned int* qr = (unsigned int*)(q + row * (long)K);
#pragma unroll
  for (int i = 0; i < 8; ++i)
    qr[i * 32 + lane] = pack_fp8x4(v[i].x * inv, v[i].y * inv,
                                   v[i].z * inv, v[i].w * inv);
}

__global__ __launch_bounds__(256) void fp8_wmma_gemm(
    const unsigned char* __restrict__ qa,
    const unsigned char* __restrict__ wb,
    const float* __restrict__ xscale,
    const float* __restrict__ wscale,
    const float* __restrict__ bias,
    float* __restrict__ out, int M, int N, int K)
{
  const int lane  = threadIdx.x & 31;
  const int wave  = threadIdx.x >> 5;
  const int waveM = wave >> 2;
  const int waveN = wave & 3;
  const int m0 = blockIdx.y * 64 + waveM * 32;
  const int n0 = blockIdx.x * 256 + waveN * 64;
  const int lr = lane & 15;
  const int lh = lane >> 4;

  const unsigned char* pa0 = qa + (long)(m0 + lr) * K + lh * 16;
  const unsigned char* pa1 = pa0 + (long)16 * K;
  const unsigned char* pb0 = wb + (long)(n0 + lr) * K + lh * 16;

  v8f c[2][4] = {};

  for (int k0 = 0; k0 < K; k0 += 128) {
    if (k0 + 128 < K) {
      __builtin_prefetch(pa0 + k0 + 128, 0, 3);
      __builtin_prefetch(pa1 + k0 + 128, 0, 3);
      __builtin_prefetch(pb0 + k0 + 128, 0, 3);
    }

    v16i a[2], b[4];
#pragma unroll
    for (int i = 0; i < 2; ++i) {
      const unsigned char* p = (i ? pa1 : pa0) + k0;
      v2i d0 = *(const v2i*)(p +   0);
      v2i d1 = *(const v2i*)(p +   8);
      v2i d2 = *(const v2i*)(p +  32);
      v2i d3 = *(const v2i*)(p +  40);
      v2i d4 = *(const v2i*)(p +  64);
      v2i d5 = *(const v2i*)(p +  72);
      v2i d6 = *(const v2i*)(p +  96);
      v2i d7 = *(const v2i*)(p + 104);
      v4i q01 = __builtin_shufflevector(d0, d1, 0, 1, 2, 3);
      v4i q23 = __builtin_shufflevector(d2, d3, 0, 1, 2, 3);
      v4i q45 = __builtin_shufflevector(d4, d5, 0, 1, 2, 3);
      v4i q67 = __builtin_shufflevector(d6, d7, 0, 1, 2, 3);
      v8i h0 = __builtin_shufflevector(q01, q23, 0, 1, 2, 3, 4, 5, 6, 7);
      v8i h1 = __builtin_shufflevector(q45, q67, 0, 1, 2, 3, 4, 5, 6, 7);
      a[i] = __builtin_shufflevector(h0, h1, 0, 1, 2, 3, 4, 5, 6, 7,
                                             8, 9, 10, 11, 12, 13, 14, 15);
    }
#pragma unroll
    for (int j = 0; j < 4; ++j) {
      const unsigned char* p = pb0 + (long)(16 * j) * K + k0;
      v4i t0 = *(const v4i*)(p +  0);
      v4i t1 = *(const v4i*)(p + 32);
      v4i t2 = *(const v4i*)(p + 64);
      v4i t3 = *(const v4i*)(p + 96);
      v8i h0 = __builtin_shufflevector(t0, t1, 0, 1, 2, 3, 4, 5, 6, 7);
      v8i h1 = __builtin_shufflevector(t2, t3, 0, 1, 2, 3, 4, 5, 6, 7);
      b[j] = __builtin_shufflevector(h0, h1, 0, 1, 2, 3, 4, 5, 6, 7,
                                             8, 9, 10, 11, 12, 13, 14, 15);
    }

#pragma unroll
    for (int i = 0; i < 2; ++i)
#pragma unroll
      for (int j = 0; j < 4; ++j)
        c[i][j] = __builtin_amdgcn_wmma_f32_16x16x128_fp8_fp8(
            a[i], b[j], (short)0, c[i][j], false, false);
  }

  const float ws = *wscale;
  float srow[2][8];
#pragma unroll
  for (int i = 0; i < 2; ++i)
#pragma unroll
    for (int v = 0; v < 8; ++v)
      srow[i][v] = xscale[m0 + i * 16 + lh * 8 + v] * ws;

#pragma unroll
  for (int i = 0; i < 2; ++i) {
#pragma unroll
    for (int j = 0; j < 4; ++j) {
      const int n = n0 + 16 * j + lr;
      const float bn = bias[n];
#pragma unroll
      for (int v = 0; v < 8; ++v) {
        const int m = m0 + i * 16 + lh * 8 + v;
        out[(long)m * N + n] = c[i][j][v] * srow[i][v] + bn;
      }
    }
  }
}

extern "C" void kernel_launch(void* const* d_in, const int* in_sizes, int n_in,
                              void* d_out, int out_size, void* d_ws, size_t ws_size,
                              hipStream_t stream) {
  const float*         x      = (const float*)d_in[0];
  const unsigned char* weight = (const unsigned char*)d_in[1];
  const float*         wscale = (const float*)d_in[2];
  const float*         bias   = (const float*)d_in[3];
  float*               out    = (float*)d_out;

  const int N = in_sizes[3];
  const int K = in_sizes[1] / N;
  const int M = in_sizes[0] / K;

  float*         xscale = (float*)d_ws;
  unsigned char* q      = (unsigned char*)d_ws + (size_t)M * 4;

  fp8_quant_rows<<<M / 8, 256, 0, stream>>>(x, q, xscale, K);

  dim3 grid((N + 255) / 256, (M + 63) / 64);
  fp8_wmma_gemm<<<grid, 256, 0, stream>>>(q, weight, xscale, wscale, bias,
                                          out, M, N, K);
}
